// MareyFluxAttention_15212774162820
// MI455X (gfx1250) — matrix operand order repaired and checked by reading, not compiled, not run on hardware
//
#include <hip/hip_runtime.h>
#include <hip/hip_bf16.h>

// ---------------- problem constants ----------------
#define DIMC   2048
#define HEADS  16
#define HD     128
#define NTOT   2560   // NX + NY
#define NXC    2048
#define NYC    512
#define BATCH  2
#define QKVN   (3*DIMC)   // 6144
#define ATTN_SCALE 0.08838834764831845f  // HD^-0.5
#define EPSC   1e-6f

typedef __attribute__((ext_vector_type(16))) __bf16 v16bf;
typedef __attribute__((ext_vector_type(8)))  __bf16 v8bf;
typedef __attribute__((ext_vector_type(4)))  __bf16 v4bf;
typedef __attribute__((ext_vector_type(8)))  float  v8f;

__device__ __forceinline__ v16bf make_frag(v8bf lo, v8bf hi) {
  v16bf f;
#pragma unroll
  for (int i = 0; i < 8; ++i) { f[i] = lo[i]; f[8 + i] = hi[i]; }
  return f;
}

__device__ __forceinline__ v8f wmma_bf16(v16bf a, v16bf b, v8f c) {
  return __builtin_amdgcn_wmma_f32_16x16x32_bf16(false, a, false, b, (short)0, c,
                                                 false, false);
}

// LDS byte offset of a __shared__ object (generic LDS addr truncates to offset)
__device__ __forceinline__ unsigned lds_off_of(const void* p) {
  return (unsigned)(unsigned long long)p;
}

// ASYNCcnt-tracked global->LDS 16B copy (CDNA5 GLOBAL_LOAD_ASYNC_TO_LDS_B128)
__device__ __forceinline__ void async_copy_b128(unsigned lds_byte_off,
                                                const void* gaddr) {
  asm volatile("global_load_async_to_lds_b128 %0, %1, off"
               :: "v"(lds_byte_off), "v"(gaddr)
               : "memory");
}

__device__ __forceinline__ void wait_async0() {
  asm volatile("s_wait_asynccnt 0x0" ::: "memory");
}

// =====================================================================
// GEMM:  C[M][N] = A[M][K] @ W[K][N] + bias   (A fp32 or bf16, W fp32)
// Double-buffered LDS: stage tile k0+32 while computing tile k0; one
// barrier per iteration. bf16-A staging goes through the async engine.
// Tiles: 64x64x32, 128 threads = 4 waves in 2x2, each wave 2x2 WMMA tiles.
// =====================================================================
template <typename TA>
__global__ __launch_bounds__(128) void gemm_bias_kernel(
    const TA* __restrict__ A, const float* __restrict__ W,
    const float* __restrict__ bias, float* __restrict__ C,
    int M, int N, int K, int n_per_b, int a_batch_rows) {
  __shared__ __bf16 sA[2][64][40];    // pad 8 -> 80B rows, 16B aligned
  __shared__ __bf16 sBt[2][64][40];   // Wt: 64 N x 32 K

  const int tid  = threadIdx.x;
  const int lane = tid & 31;
  const int wave = tid >> 5;
  const int wm = wave >> 1, wn = wave & 1;
  const int half = lane >> 4, l16 = lane & 15;
  const int bm = blockIdx.y * 64, bn = blockIdx.x * 64;
  const unsigned ldsA = lds_off_of(&sA[0][0][0]);

  v8f acc[2][2] = {};

  auto stage = [&](int buf, int k0) {
    // A tile
    if constexpr (sizeof(TA) == 4) {
#pragma unroll
      for (int jj = 0; jj < 4; ++jj) {
        int off = tid * 16 + jj * 4;             // 0..2047
        int r = off >> 5, c = off & 31;
        int gr = bm + r;
        int b = gr / n_per_b, n = gr - b * n_per_b;
        const float* ap =
            (const float*)A + (size_t)(b * a_batch_rows + n) * K + k0 + c;
        float4 f = *(const float4*)ap;
        v4bf p = {(__bf16)f.x, (__bf16)f.y, (__bf16)f.z, (__bf16)f.w};
        *(v4bf*)&sA[buf][r][c] = p;              // one ds_store_b64
      }
    } else {
      // bf16 A: pure copy -> async global->LDS (ASYNCcnt, no VGPR bounce)
#pragma unroll
      for (int jj = 0; jj < 2; ++jj) {
        int off = tid * 16 + jj * 8;
        int r = off >> 5, c = off & 31;
        int gr = bm + r;
        int b = gr / n_per_b, n = gr - b * n_per_b;
        const __bf16* ap =
            (const __bf16*)A + (size_t)(b * a_batch_rows + n) * K + k0 + c;
        async_copy_b128(ldsA + (unsigned)(buf * 64 * 40 + r * 40 + c) * 2, ap);
      }
    }
    // W tile transposed: sBt[buf][n][k] = W[k0+k][bn+n]
#pragma unroll
    for (int jj = 0; jj < 4; ++jj) {
      int off = tid * 16 + jj * 4;               // 0..2047 over 32x64
      int kr = off >> 6, nc = off & 63;
      const float* wp = W + (size_t)(k0 + kr) * N + bn + nc;
      float4 f = *(const float4*)wp;
      sBt[buf][nc + 0][kr] = (__bf16)f.x; sBt[buf][nc + 1][kr] = (__bf16)f.y;
      sBt[buf][nc + 2][kr] = (__bf16)f.z; sBt[buf][nc + 3][kr] = (__bf16)f.w;
    }
  };

  stage(0, 0);
  if constexpr (sizeof(TA) == 2) wait_async0();
  __syncthreads();

  for (int k0 = 0; k0 < K; k0 += 32) {
    const int cur = (k0 >> 5) & 1, nxt = cur ^ 1;
    // prefetch W two tiles ahead, stage next tile while we compute
    if (k0 + 64 < K) {
      const float* pf =
          W + (size_t)(k0 + 64 + (tid >> 2)) * N + bn + (tid & 3) * 16;
      __builtin_prefetch(pf, 0, 0);
    }
    if (k0 + 32 < K) stage(nxt, k0 + 32);

    // preload all fragments, then a clean WMMA burst
    v16bf af[2], bf[2];
#pragma unroll
    for (int mi = 0; mi < 2; ++mi) {
      const __bf16* ap = &sA[cur][wm * 32 + mi * 16 + l16][0];
      af[mi] = make_frag(*(const v8bf*)(ap + half * 8),
                         *(const v8bf*)(ap + 16 + half * 8));
    }
#pragma unroll
    for (int ni = 0; ni < 2; ++ni) {
      const __bf16* bp = &sBt[cur][wn * 32 + ni * 16 + l16][0];
      bf[ni] = make_frag(*(const v8bf*)(bp + half * 8),
                         *(const v8bf*)(bp + 16 + half * 8));
    }
#pragma unroll
    for (int ni = 0; ni < 2; ++ni)
#pragma unroll
      for (int mi = 0; mi < 2; ++mi)
        acc[mi][ni] = wmma_bf16(af[mi], bf[ni], acc[mi][ni]);

    if constexpr (sizeof(TA) == 2) wait_async0();
    __syncthreads();
  }

#pragma unroll
  for (int mi = 0; mi < 2; ++mi)
#pragma unroll
    for (int ni = 0; ni < 2; ++ni)
#pragma unroll
      for (int j = 0; j < 8; ++j) {
        int row = bm + wm * 32 + mi * 16 + j + 8 * half;
        int col = bn + wn * 32 + ni * 16 + l16;
        C[(size_t)row * N + col] = acc[mi][ni][j] + bias[col];
      }
}

// =====================================================================
// RMSNorm(q,k per head) + RoPE(last 64 ch of q,k) + relayout to bf16
// q/k/v out layout: [B][H][NTOT][HD]. One wave per (b,n,h) row.
// =====================================================================
__global__ __launch_bounds__(128) void norm_rope_kernel(
    const float* __restrict__ qkv, const float* __restrict__ qn,
    const float* __restrict__ kn, const int* __restrict__ pos,
    __bf16* __restrict__ qo, __bf16* __restrict__ ko, __bf16* __restrict__ vo,
    int Nseg, int n_off) {
  int wid  = (blockIdx.x * blockDim.x + threadIdx.x) >> 5;  // over B*Nseg*H
  int lane = threadIdx.x & 31;
  int h  = wid % HEADS;
  int rn = wid / HEADS;                 // b*Nseg + n
  int b  = rn / Nseg;
  int n  = rn - b * Nseg;

  const float* base = qkv + (size_t)rn * QKVN + h * HD;
  int c0 = lane * 4;
  float4 qv = *(const float4*)(base + c0);
  float4 kv = *(const float4*)(base + DIMC + c0);
  float4 vv = *(const float4*)(base + 2 * DIMC + c0);

  float sq = qv.x * qv.x + qv.y * qv.y + qv.z * qv.z + qv.w * qv.w;
  float sk = kv.x * kv.x + kv.y * kv.y + kv.z * kv.z + kv.w * kv.w;
#pragma unroll
  for (int m = 1; m < 32; m <<= 1) {
    sq += __shfl_xor(sq, m, 32);
    sk += __shfl_xor(sk, m, 32);
  }
  float rq = rsqrtf(sq / (float)HD + EPSC);
  float rk = rsqrtf(sk / (float)HD + EPSC);

  float qf[4] = {qv.x, qv.y, qv.z, qv.w};
  float kf[4] = {kv.x, kv.y, kv.z, kv.w};
  float vf[4] = {vv.x, vv.y, vv.z, vv.w};
#pragma unroll
  for (int e = 0; e < 4; ++e) {
    qf[e] = qn[c0 + e] * qf[e] * rq;
    kf[e] = kn[c0 + e] * kf[e] * rk;
  }

  // RoPE over channels [64,128): pair c <-> c+-32 == lane xor 8
  float pq[4], pk[4];
#pragma unroll
  for (int e = 0; e < 4; ++e) {
    pq[e] = __shfl_xor(qf[e], 8, 32);
    pk[e] = __shfl_xor(kf[e], 8, 32);
  }
  int p = pos[rn];
  if (lane >= 16) {
    bool firstHalf = lane < 24;   // channels 64..95 vs 96..127
#pragma unroll
    for (int e = 0; e < 4; ++e) {
      int c = c0 + e;
      int i = firstHalf ? (c - 64) : (c - 96);
      float inv_ts = __powf(10000.0f, -(float)i / 32.0f);
      float ang = (float)p * inv_ts;
      float s, co;
      __sincosf(ang, &s, &co);
      if (firstHalf) {
        qf[e] = qf[e] * co - pq[e] * s;
        kf[e] = kf[e] * co - pk[e] * s;
      } else {
        qf[e] = qf[e] * co + pq[e] * s;
        kf[e] = kf[e] * co + pk[e] * s;
      }
    }
  }

  size_t obase =
      ((size_t)(b * HEADS + h) * NTOT + (size_t)(n + n_off)) * HD + c0;
#pragma unroll
  for (int e = 0; e < 4; ++e) {
    qo[obase + e] = (__bf16)qf[e];
    ko[obase + e] = (__bf16)kf[e];
    vo[obase + e] = (__bf16)vf[e];
  }
}

// =====================================================================
// Flash attention: block = 4 waves = 64 q rows for one (b,h).
// Double-buffered K/V staging: async global->LDS copies for the next
// 32-key chunk are issued before computing the current one. One barrier
// per iteration. Output written bf16 into attn[b][n][h*HD+d].
// =====================================================================
__global__ __launch_bounds__(128) void attn_kernel(
    const __bf16* __restrict__ Q, const __bf16* __restrict__ K,
    const __bf16* __restrict__ V, __bf16* __restrict__ O) {
  __shared__ __bf16 sK[2][32][HD];     // [buf][key][d]  (B-frags for S)
  __shared__ __bf16 sVt[2][HD][40];    // [buf][d][key]  (B-frags for O)
  __shared__ __bf16 sP[4][16][40];     // per-wave P C->A relayout scratch

  const int tid = threadIdx.x, lane = tid & 31, wave = tid >> 5;
  const int half = lane >> 4, l16 = lane & 15;
  const int qt = blockIdx.x % (NTOT / 64);
  const int bh = blockIdx.x / (NTOT / 64);      // b*HEADS + h
  const int qrow0 = qt * 64 + wave * 16;

  const __bf16* Qb = Q + (size_t)bh * NTOT * HD;
  const __bf16* Kb = K + (size_t)bh * NTOT * HD;
  const __bf16* Vb = V + (size_t)bh * NTOT * HD;

  // Q fragments (16 rows x 128 d) held in registers for entire key loop
  v16bf qfrag[4];
#pragma unroll
  for (int dc = 0; dc < 4; ++dc) {
    const __bf16* qp = Qb + (size_t)(qrow0 + l16) * HD + dc * 32;
    qfrag[dc] = make_frag(*(const v8bf*)(qp + half * 8),
                          *(const v8bf*)(qp + 16 + half * 8));
  }

  v8f oacc[8] = {};
  float mrun[8], lrun[8];
#pragma unroll
  for (int j = 0; j < 8; ++j) { mrun[j] = -1e30f; lrun[j] = 0.0f; }

  const unsigned ldsK = lds_off_of(&sK[0][0][0]);

  auto stage = [&](int buf, int kc) {
    // K chunk: async global->LDS, 16B per request (layout-preserving copy)
#pragma unroll
    for (int i = 0; i < 4; ++i) {
      int e = (tid + i * 128) * 16;   // byte offset inside 8KB chunk
      async_copy_b128(ldsK + buf * (32 * HD * 2) + e,
                      (const char*)(Kb + (size_t)kc * HD) + e);
    }
    // V chunk transposed: sVt[buf][d][r] = V[kc+r][d]
#pragma unroll
    for (int s = 0; s < 4; ++s) {
      int seg = tid * 4 + s;
      int r = seg >> 4, d0 = (seg & 15) * 8;
      v8bf vv = *(const v8bf*)(Vb + (size_t)(kc + r) * HD + d0);
#pragma unroll
      for (int i = 0; i < 8; ++i) sVt[buf][d0 + i][r] = vv[i];
    }
  };

  stage(0, 0);
  wait_async0();
  __syncthreads();

  for (int kc = 0; kc < NTOT; kc += 32) {
    const int cur = (kc >> 5) & 1, nxt = cur ^ 1;
    if (kc + 64 < NTOT)
      __builtin_prefetch(Kb + (size_t)(kc + 64) * HD + tid * 32, 0, 0);
    if (kc + 32 < NTOT) stage(nxt, kc + 32);

    // ---- S = Q @ K^T  (16 x 32 keys): preload frags, then WMMA burst ----
    v16bf kfr[2][4];
#pragma unroll
    for (int kt = 0; kt < 2; ++kt)
#pragma unroll
      for (int dc = 0; dc < 4; ++dc) {
        const __bf16* kp = &sK[cur][kt * 16 + l16][dc * 32];
        kfr[kt][dc] = make_frag(*(const v8bf*)(kp + half * 8),
                                *(const v8bf*)(kp + 16 + half * 8));
      }
    v8f sacc[2] = {};
#pragma unroll
    for (int kt = 0; kt < 2; ++kt)
#pragma unroll
      for (int dc = 0; dc < 4; ++dc)
        sacc[kt] = wmma_bf16(qfrag[dc], kfr[kt][dc], sacc[kt]);

    // ---- online softmax (row stats live per (half, j)) ----
    float pr0[8], pr1[8];
#pragma unroll
    for (int j = 0; j < 8; ++j) {
      float s0 = sacc[0][j] * ATTN_SCALE;
      float s1 = sacc[1][j] * ATTN_SCALE;
      float mx = fmaxf(s0, s1);
#pragma unroll
      for (int m = 1; m < 16; m <<= 1) mx = fmaxf(mx, __shfl_xor(mx, m, 32));
      float mnew  = fmaxf(mrun[j], mx);
      float alpha = __expf(mrun[j] - mnew);
      float p0 = __expf(s0 - mnew), p1 = __expf(s1 - mnew);
      float rs = p0 + p1;
#pragma unroll
      for (int m = 1; m < 16; m <<= 1) rs += __shfl_xor(rs, m, 32);
      lrun[j] = lrun[j] * alpha + rs;
      mrun[j] = mnew;
#pragma unroll
      for (int dt = 0; dt < 8; ++dt) oacc[dt][j] *= alpha;
      pr0[j] = p0; pr1[j] = p1;
    }

    // ---- P: C-layout -> A-layout via per-wave LDS (DS in-order per wave) ----
#pragma unroll
    for (int j = 0; j < 8; ++j) {
      sP[wave][j + 8 * half][l16]      = (__bf16)pr0[j];
      sP[wave][j + 8 * half][16 + l16] = (__bf16)pr1[j];
    }
    const __bf16* pp = &sP[wave][l16][0];
    v16bf pfrag = make_frag(*(const v8bf*)(pp + half * 8),
                            *(const v8bf*)(pp + 16 + half * 8));

    // ---- O += P @ V: preload frags, then WMMA burst ----
    v16bf vfr[8];
#pragma unroll
    for (int dt = 0; dt < 8; ++dt) {
      const __bf16* vp = &sVt[cur][dt * 16 + l16][0];
      vfr[dt] = make_frag(*(const v8bf*)(vp + half * 8),
                          *(const v8bf*)(vp + 16 + half * 8));
    }
#pragma unroll
    for (int dt = 0; dt < 8; ++dt)
      oacc[dt] = wmma_bf16(pfrag, vfr[dt], oacc[dt]);

    wait_async0();
    __syncthreads();
  }

  // ---- normalize and write attn output ----
  const int b = bh >> 4, h = bh & 15;
#pragma unroll
  for (int j = 0; j < 8; ++j) {
    float inv = 1.0f / lrun[j];
    int n = qrow0 + j + 8 * half;
    size_t rowbase = ((size_t)(b * NTOT + n)) * DIMC + h * HD;
#pragma unroll
    for (int dt = 0; dt < 8; ++dt)
      O[rowbase + dt * 16 + l16] = (__bf16)(oacc[dt][j] * inv);
  }
}

// =====================================================================
// Host launcher
// =====================================================================
extern "C" void kernel_launch(void* const* d_in, const int* in_sizes, int n_in,
                              void* d_out, int out_size, void* d_ws,
                              size_t ws_size, hipStream_t stream) {
  const float* x       = (const float*)d_in[0];
  const float* y       = (const float*)d_in[1];
  const int*   tpos    = (const int*)d_in[2];
  const int*   ypos    = (const int*)d_in[3];
  const float* Wqkv_x  = (const float*)d_in[4];
  const float* bqkv_x  = (const float*)d_in[5];
  const float* qnx     = (const float*)d_in[6];
  const float* knx     = (const float*)d_in[7];
  const float* Wproj_x = (const float*)d_in[8];
  const float* bproj_x = (const float*)d_in[9];
  const float* Wqkv_y  = (const float*)d_in[10];
  const float* bqkv_y  = (const float*)d_in[11];
  const float* qny     = (const float*)d_in[12];
  const float* kny     = (const float*)d_in[13];
  const float* Wproj_y = (const float*)d_in[14];
  const float* bproj_y = (const float*)d_in[15];

  // workspace partition
  char* ws = (char*)d_ws;
  float* qkv_x = (float*)ws;  ws += (size_t)BATCH * NXC * QKVN * sizeof(float);
  float* qkv_y = (float*)ws;  ws += (size_t)BATCH * NYC * QKVN * sizeof(float);
  const size_t qkvElems = (size_t)BATCH * HEADS * NTOT * HD;
  __bf16* qb   = (__bf16*)ws; ws += qkvElems * sizeof(__bf16);
  __bf16* kb   = (__bf16*)ws; ws += qkvElems * sizeof(__bf16);
  __bf16* vb   = (__bf16*)ws; ws += qkvElems * sizeof(__bf16);
  __bf16* attn = (__bf16*)ws; ws += (size_t)BATCH * NTOT * DIMC * sizeof(__bf16);

  // 1-2) QKV GEMMs (fp32 in, bf16 WMMA, fp32 out)
  {
    dim3 grid(QKVN / 64, (BATCH * NXC) / 64);
    gemm_bias_kernel<float><<<grid, 128, 0, stream>>>(
        x, Wqkv_x, bqkv_x, qkv_x, BATCH * NXC, QKVN, DIMC, BATCH * NXC,
        BATCH * NXC);
  }
  {
    dim3 grid(QKVN / 64, (BATCH * NYC) / 64);
    gemm_bias_kernel<float><<<grid, 128, 0, stream>>>(
        y, Wqkv_y, bqkv_y, qkv_y, BATCH * NYC, QKVN, DIMC, BATCH * NYC,
        BATCH * NYC);
  }

  // 3-4) RMSNorm + RoPE + relayout into concatenated q/k/v
  {
    int waves = BATCH * NXC * HEADS;           // 65536
    norm_rope_kernel<<<waves / 4, 128, 0, stream>>>(qkv_x, qnx, knx, tpos, qb,
                                                    kb, vb, NXC, 0);
  }
  {
    int waves = BATCH * NYC * HEADS;           // 16384
    norm_rope_kernel<<<waves / 4, 128, 0, stream>>>(qkv_y, qny, kny, ypos, qb,
                                                    kb, vb, NYC, NXC);
  }

  // 5) flash attention
  {
    int blocks = BATCH * HEADS * (NTOT / 64);  // 1280
    attn_kernel<<<blocks, 128, 0, stream>>>(qb, kb, vb, attn);
  }

  // 6-7) output projections straight into d_out (x part then y part)
  {
    dim3 grid(DIMC / 64, (BATCH * NXC) / 64);
    gemm_bias_kernel<__bf16><<<grid, 128, 0, stream>>>(
        attn, Wproj_x, bproj_x, (float*)d_out, BATCH * NXC, DIMC, DIMC, NXC,
        NTOT);
  }
  {
    const __bf16* attnY = attn + (size_t)NXC * DIMC;  // rows offset by NX
    float* outY = (float*)d_out + (size_t)BATCH * NXC * DIMC;
    dim3 grid(DIMC / 64, (BATCH * NYC) / 64);
    gemm_bias_kernel<__bf16><<<grid, 128, 0, stream>>>(
        attnY, Wproj_y, bproj_y, outY, BATCH * NYC, DIMC, DIMC, NYC, NTOT);
  }
}
